// ModDemodConv3x3_59081570125043
// MI455X (gfx1250) — matrix operand order repaired and checked by reading, not compiled, not run on hardware
//
#include <hip/hip_runtime.h>
#include <hip/hip_bf16.h>

// ModDemodConv3x3 on MI455X (gfx1250, wave32, WMMA + TDM):
//   k1 : modulate+demodulate -> bf16 weights, chunk-major wmod[b][kc][cout][tap][cl]
//   k1b: x f32 NCHW -> bf16 channels-last x_t[b][y][x][cin]
//   k2 : implicit GEMM conv; per 32-cin chunk, stage A(64x9x32) + X(3 rows halo)
//        via TENSOR_LOAD_TO_LDS (TDM, with LDS padding for bank stagger), then
//        9 taps x 2 WMMA tiles/wave out of LDS (18 wmma / 2 barriers / chunk).

typedef __attribute__((ext_vector_type(16))) __bf16 v16bf;
typedef __attribute__((ext_vector_type(8)))  __bf16 v8bf;
typedef __attribute__((ext_vector_type(8)))  float  v8f;
typedef __attribute__((ext_vector_type(4)))  unsigned int u32x4;
typedef __attribute__((ext_vector_type(8)))  int          i32x8;
typedef __attribute__((ext_vector_type(4)))  int          i32x4;

#define CIN    512
#define COUT   512
#define HW     4096            // 64*64
#define NCHUNK 16              // 512/32 cin chunks
#define CROW   288             // 9 taps * 32 cins per cout row per chunk (global)
#define ATAPP  40              // LDS A per-tap pitch (elems): 32 data + 4DW pad
#define AROWP  360             // LDS A per-cout pitch: 9*40 (720B, 16B aligned)
#define XPITCH 40              // LDS X per-pixel pitch (elems): 80B, 16B aligned

#define WMOD_BYTES ((size_t)8 * NCHUNK * COUT * CROW * 2)   // 36 MiB

#if __has_builtin(__builtin_amdgcn_tensor_load_to_lds) && \
    __has_builtin(__builtin_amdgcn_s_wait_tensorcnt)
#define USE_TDM 1
#else
#define USE_TDM 0
#endif

// ---------------------------------------------------------------------------
// Kernel 1: per-(b,cout) modulation + demodulation, bf16, chunk-major layout.
// ---------------------------------------------------------------------------
__global__ __launch_bounds__(256)
void moddemod_kernel(const float* __restrict__ style,
                     const float* __restrict__ weight,
                     __bf16* __restrict__ wmod)
{
    const int bc   = blockIdx.x;
    const int b    = bc >> 9;
    const int cout = bc & 511;
    const int t    = threadIdx.x;

    float vals[18];
    float ss = 0.f;
#pragma unroll
    for (int i = 0; i < 18; ++i) {
        const int e   = t + i * 256;      // e = tap*512 + cin
        const int tap = e >> 9;
        const int cin = e & 511;
        const float v = style[b * CIN + cin] *
                        weight[(cout * CIN + cin) * 9 + tap];
        vals[i] = v;
        ss += v * v;
    }

    __shared__ float red[256];
    red[t] = ss;
    __syncthreads();
#pragma unroll
    for (int s = 128; s > 0; s >>= 1) {
        if (t < s) red[t] += red[t + s];
        __syncthreads();
    }
    const float inv = rsqrtf(red[0] + 0.001f);

#pragma unroll
    for (int i = 0; i < 18; ++i) {
        const int e   = t + i * 256;
        const int tap = e >> 9;
        const int cin = e & 511;
        const int kc  = cin >> 5;
        const int cl  = cin & 31;
        wmod[((size_t)(b * NCHUNK + kc) * COUT + cout) * CROW + tap * 32 + cl] =
            (__bf16)(vals[i] * inv);
    }
}

// ---------------------------------------------------------------------------
// Kernel 1b: x[b][cin][y][xc] f32  ->  x_t[b][y][xc][cin] bf16 (LDS transpose)
// ---------------------------------------------------------------------------
__global__ __launch_bounds__(256)
void convert_x_kernel(const float* __restrict__ x, __bf16* __restrict__ xt)
{
    const int y = blockIdx.x;
    const int b = blockIdx.y;
    const int t = threadIdx.x;

    const float* xs = x  + (size_t)b * CIN * HW + y * 64;
    __bf16*      xd = xt + (size_t)(b * 64 + y) * 64 * CIN;

    __shared__ __bf16 tile[32 * 72];

    const int xc  = t & 63;
    const int cg  = t >> 6;
    const int cl2 = t & 31;
    const int xg  = t >> 5;

    for (int chunk = 0; chunk < NCHUNK; ++chunk) {
        const int cin0 = chunk << 5;
#pragma unroll
        for (int i = 0; i < 8; ++i) {
            const int cl = (cg << 3) + i;
            tile[cl * 72 + xc] = (__bf16)xs[(size_t)(cin0 + cl) * HW + xc];
        }
        __syncthreads();
#pragma unroll
        for (int i = 0; i < 8; ++i) {
            const int xc2 = (xg << 3) + i;
            xd[(size_t)xc2 * CIN + cin0 + cl2] = tile[cl2 * 72 + xc2];
        }
        __syncthreads();
    }
}

// ---------------------------------------------------------------------------
// TDM descriptor packing (CDNA5 ISA).
// Group0: count=1, lds_addr, 57b global_addr, type=2.
// Group1: data_size=2B, pad_enable, pad_interval=3 (16 DW), pad_amount=3 (4 DW)
//         -> every 64B of data followed by 16B LDS skip: 32-elem rows on a
//            40-elem pitch, matching the bank-staggered LDS layouts below.
// ---------------------------------------------------------------------------
#if USE_TDM
__device__ __forceinline__ u32x4 tdm_g0(unsigned lds_off, unsigned long long ga)
{
    u32x4 g;
    g.x = 1u;                                                   // count=1
    g.y = lds_off;                                              // lds_addr
    g.z = (unsigned)ga;                                         // global_addr lo
    g.w = ((unsigned)(ga >> 32) & 0x01FFFFFFu) | 0x80000000u;   // hi | type=2
    return g;
}

__device__ __forceinline__ i32x8 tdm_g1(unsigned dim0, unsigned dim1,
                                        unsigned tile0, unsigned tile1,
                                        unsigned stride0)
{
    i32x8 g;
    g[0] = (int)((1u << 16) | (1u << 20) | (3u << 22) | (3u << 25));
    g[1] = (int)(dim0 << 16);                       // tensor_dim0[15:0]
    g[2] = (int)((dim0 >> 16) | (dim1 << 16));      // dim0 hi | dim1 lo
    g[3] = (int)((dim1 >> 16) | (tile0 << 16));     // dim1 hi | tile_dim0
    g[4] = (int)(tile1 & 0xFFFFu);                  // tile_dim1 (tile_dim2=0)
    g[5] = (int)stride0;                            // tensor_dim0_stride lo
    g[6] = 0;
    g[7] = 0;
    return g;
}
#endif

// ---------------------------------------------------------------------------
// Kernel 2: implicit GEMM conv via v_wmma_f32_16x16x32_bf16.
//   grid = (64 y, 8 coutTiles, 8 b), block = 256 (8 waves).
// ---------------------------------------------------------------------------
__global__ __launch_bounds__(256)
void conv_wmma_kernel(const __bf16* __restrict__ xt,
                      const __bf16* __restrict__ wmod,
                      float* __restrict__ out)
{
    const int y        = blockIdx.x;
    const int coutBase = blockIdx.y << 6;
    const int b        = blockIdx.z;

    const int t    = threadIdx.x;
    const int lane = t & 31;
    const int wave = t >> 5;
    const int mw   = wave & 3;
    const int nw   = wave >> 2;

    __shared__ __bf16 As[64 * AROWP];         // [cout 64][tap 9][cl 32 (+pad)]
    __shared__ __bf16 Xs[3 * 66 * XPITCH];    // [row 3][px 66 halo][cl 32 (+pad)]

    v8f acc0 = {};
    v8f acc1 = {};

    const __bf16* xtb = xt + (size_t)b * HW * CIN;    // [y][x][cin]

    // fragment addressing (ISA 7.12.2 bf16 layouts, wave32)
    const int ar  = mw * 16 + (lane & 15);
    const int akb = (lane >> 4) << 3;         // A K base: 0 or 8 (+16 for hi)
    const int kh  = (lane >> 4) << 3;         // B K base: 0 or 8 (+16 for hi)
    const int n0  = nw * 32 + (lane & 15);
    const int n1  = n0 + 16;

    // ---- one-time zeroing: column halo (px 0, 65) and OOB image rows
    {
        v8bf z = {};
        if (t < 24) {
            const int row = t >> 3;
            const int rem = t & 7;
            const int px  = (rem >> 2) ? 65 : 0;
            *(v8bf*)(Xs + (row * 66 + px) * XPITCH + ((rem & 3) << 3)) = z;
        }
#if USE_TDM
        // rows y-1 / y+1 outside the image are never TDM-written: zero once.
#pragma unroll
        for (int row = 0; row < 3; ++row) {
            if ((unsigned)(y + row - 1) >= 64u) {
                const int px = t >> 2;
                const int q  = (t & 3) << 3;
                *(v8bf*)(Xs + (row * 66 + 1 + px) * XPITCH + q) = z;
            }
        }
#endif
    }
    __syncthreads();

#if USE_TDM
    const unsigned asOff = (unsigned)(size_t)(void*)As;
    const unsigned xsOff = (unsigned)(size_t)(void*)Xs;
    const i32x8 gA = tdm_g1(/*dim0=*/CROW, /*dim1=*/64,
                            /*tile0=*/CROW, /*tile1=*/64, /*stride0=*/CROW);
    const i32x8 gX = tdm_g1(/*dim0=*/32,   /*dim1=*/64,
                            /*tile0=*/32,   /*tile1=*/64, /*stride0=*/CIN);
    const i32x4 z4 = {};
    const i32x8 z8 = {};
#endif

#pragma unroll 1
    for (int kc = 0; kc < NCHUNK; ++kc) {
        const int cin0 = kc << 5;
        const __bf16* wkc =
            wmod + ((size_t)(b * NCHUNK + kc) * COUT + coutBase) * CROW;

#if USE_TDM
        // ---- stage via Tensor Data Mover (wave 0 issues; TENSORcnt tracked)
        if (wave == 0) {
            __builtin_amdgcn_tensor_load_to_lds(
                tdm_g0(asOff, (unsigned long long)(size_t)wkc),
                gA, z4, z4, z8, 0);
#pragma unroll
            for (int row = 0; row < 3; ++row) {
                const int ysrc = y + row - 1;
                if ((unsigned)ysrc < 64u) {
                    const __bf16* src =
                        xtb + (size_t)(ysrc * 64) * CIN + cin0;
                    __builtin_amdgcn_tensor_load_to_lds(
                        tdm_g0(xsOff + (unsigned)((row * 66 + 1) * XPITCH * 2),
                               (unsigned long long)(size_t)src),
                        gX, z4, z4, z8, 0);
                }
            }
            __builtin_amdgcn_s_wait_tensorcnt(0);
        }
#else
        // ---- fallback: per-thread vector staging
        {
            const int arow = t >> 2;
            const int aq   = (t & 3) << 3;
            const __bf16* srow = wkc + (size_t)arow * CROW + aq;
            __bf16*       drow = As + arow * AROWP + aq;
#pragma unroll
            for (int tap = 0; tap < 9; ++tap)
                *(v8bf*)(drow + tap * ATAPP) = *(const v8bf*)(srow + tap * 32);
            __builtin_prefetch((const void*)(srow + (size_t)COUT * CROW), 0, 1);
        }
#pragma unroll
        for (int j = 0; j < 3; ++j) {
            const int s    = j * 256 + t;
            const int pair = s >> 2;
            const int q    = (s & 3) << 3;
            const int row  = pair >> 6;
            const int px   = pair & 63;
            const int ysrc = y + row - 1;
            v8bf v = {};
            if ((unsigned)ysrc < 64u)
                v = *(const v8bf*)(xtb + (size_t)(ysrc * 64 + px) * CIN + cin0 + q);
            *(v8bf*)(Xs + (row * 66 + px + 1) * XPITCH + q) = v;
        }
#endif
        __syncthreads();

        // ---- 9 taps entirely out of LDS: 18 WMMA per chunk
#pragma unroll
        for (int tap = 0; tap < 9; ++tap) {
            const int dy = tap / 3;
            const int dx = tap % 3;

            v8bf alo = *(const v8bf*)(As + ar * AROWP + tap * ATAPP + akb);
            v8bf ahi = *(const v8bf*)(As + ar * AROWP + tap * ATAPP + akb + 16);
            v16bf afrag = __builtin_shufflevector(alo, ahi,
                0,1,2,3,4,5,6,7,8,9,10,11,12,13,14,15);

            v8bf b0lo = *(const v8bf*)(Xs + (dy * 66 + n0 + dx) * XPITCH + kh);
            v8bf b0hi = *(const v8bf*)(Xs + (dy * 66 + n0 + dx) * XPITCH + kh + 16);
            v16bf bfrag0 = __builtin_shufflevector(b0lo, b0hi,
                0,1,2,3,4,5,6,7,8,9,10,11,12,13,14,15);

            v8bf b1lo = *(const v8bf*)(Xs + (dy * 66 + n1 + dx) * XPITCH + kh);
            v8bf b1hi = *(const v8bf*)(Xs + (dy * 66 + n1 + dx) * XPITCH + kh + 16);
            v16bf bfrag1 = __builtin_shufflevector(b1lo, b1hi,
                0,1,2,3,4,5,6,7,8,9,10,11,12,13,14,15);

            acc0 = __builtin_amdgcn_wmma_f32_16x16x32_bf16(
                false, afrag, false, bfrag0, (short)0, acc0, false, false);
            acc1 = __builtin_amdgcn_wmma_f32_16x16x32_bf16(
                false, afrag, false, bfrag1, (short)0, acc1, false, false);
        }
        __syncthreads();
    }

    // ---- store: C/D layout: VGPR v -> M = v + 8*(lane/16), N = lane%16
    float* ob = out + ((size_t)(b * COUT + coutBase) * 64 + y) * 64;
    const int mrow = mw * 16 + ((lane >> 4) << 3);
    const int ncol = nw * 32 + (lane & 15);
#pragma unroll
    for (int v = 0; v < 8; ++v) {
        const int m = mrow + v;
        ob[(size_t)m * HW + ncol]      = acc0[v];
        ob[(size_t)m * HW + ncol + 16] = acc1[v];
    }
}

// ---------------------------------------------------------------------------
extern "C" void kernel_launch(void* const* d_in, const int* in_sizes, int n_in,
                              void* d_out, int out_size, void* d_ws, size_t ws_size,
                              hipStream_t stream)
{
    const float* x      = (const float*)d_in[0];  // [8,512,64,64]
    const float* style  = (const float*)d_in[1];  // [8,512]
    const float* weight = (const float*)d_in[2];  // [512,512,3,3]
    float*       out    = (float*)d_out;          // [8,512,64,64]

    __bf16* wmod = (__bf16*)d_ws;                             // 36 MiB
    __bf16* xtv  = (__bf16*)((char*)d_ws + WMOD_BYTES);       // 32 MiB

    (void)in_sizes; (void)n_in; (void)out_size; (void)ws_size;

    moddemod_kernel<<<8 * COUT, 256, 0, stream>>>(style, weight, wmod);
    convert_x_kernel<<<dim3(64, 8), 256, 0, stream>>>(x, xtv);
    conv_wmma_kernel<<<dim3(64, COUT / 64, 8), 256, 0, stream>>>(xtv, wmod, out);
}
